// Lorem_83416854823263
// MI455X (gfx1250) — matrix operand order repaired and checked by reading, not compiled, not run on hardware
//
#include <hip/hip_runtime.h>
#include <math.h>

typedef __attribute__((ext_vector_type(16))) _Float16 v16h;
typedef __attribute__((ext_vector_type(8)))  float    v8f;

#define N_NODES 8192
#define N_PAIRS 65536
#define DIM 128
#define PI_F 3.14159265358979f
#define CUT_F 5.0f

// Compiler-level fence: stops LICM from hoisting loop-invariant LDS fragment
// loads out of reduction loops (which caused scratch spills in round 2).
#define NO_HOIST() asm volatile("" ::: "memory")

__device__ __forceinline__ float silu_f(float v) { return v / (1.0f + expf(-v)); }

// ---------------------------------------------------------------------------
// Weight convert + transpose: W (K x Nc row-major f32) -> out (Nc x K, f16)
// Column-major so a WMMA B-lane (one output column) reads contiguous K.
// ---------------------------------------------------------------------------
__global__ void convT(const float* __restrict__ W, _Float16* __restrict__ outp,
                      int K, int Nc) {
  int idx = blockIdx.x * 256 + threadIdx.x;
  if (idx >= K * Nc) return;
  int k = idx / Nc, n = idx - k * Nc;
  outp[(size_t)n * K + k] = (_Float16)W[idx];
}

// ---------------------------------------------------------------------------
// Per-edge geometry + pair features + rc1 layer-1 (16->128, VALU).
// One block (128 threads) per edge.
// ---------------------------------------------------------------------------
__global__ void edge_geom(const float* __restrict__ Rij, const int* __restrict__ ii,
                          const int* __restrict__ jj, const int* __restrict__ Zi,
                          const unsigned char* __restrict__ pmask,
                          const unsigned char* __restrict__ nmask,
                          const float* __restrict__ embed,
                          const float* __restrict__ W1, const float* __restrict__ b1,
                          _Float16* __restrict__ h1, float* __restrict__ radial,
                          float* __restrict__ sphout) {
  int p = blockIdx.x;
  int t = threadIdx.x;  // 0..127
  __shared__ float sf[16];
  float rx = Rij[p * 3 + 0], ry = Rij[p * 3 + 1], rz = Rij[p * 3 + 2];
  float pm = pmask[p] ? 1.0f : 0.0f;
  float r = sqrtf(rx * rx + ry * ry + rz * rz);
  if (t < 16) {
    int node = (t < 8) ? ii[p] : jj[p];
    int c = t & 7;
    float nm = nmask[node] ? 1.0f : 0.0f;
    sf[t] = embed[Zi[node] * 8 + c] * nm;
  }
  __syncthreads();
  // h1[d] = silu(pair_feat @ W1 + b1)
  float acc = b1[t];
#pragma unroll
  for (int k = 0; k < 16; ++k) acc += sf[k] * W1[k * DIM + t];
  h1[(size_t)p * DIM + t] = (_Float16)silu_f(acc);
  // Bernstein radial basis * cosine cutoff
  if (t < 8) {
    float cut = (r < CUT_F) ? 0.5f * (cosf(PI_F * r / CUT_F) + 1.0f) : 0.0f;
    cut *= pm;
    float x = fminf(fmaxf(r / CUT_F, 1e-6f), 1.0f - 1e-6f);
    const float C[8] = {1.f, 7.f, 21.f, 35.f, 35.f, 21.f, 7.f, 1.f};
    float b = C[t] * expf((float)t * logf(x) + (float)(7 - t) * log1pf(-x));
    radial[(size_t)p * 8 + t] = b * cut;
  }
  // real spherical harmonics (l<=3), masked unit vector
  if (t < 16) {
    float inv = 1.0f / fmaxf(r, 1e-12f);
    float x = rx * inv * pm, y = ry * inv * pm, z = rz * inv * pm;
    const float s3 = 1.7320508075688772f, s15 = 3.872983346207417f;
    const float c1 = 0.7905694150420949f, c2 = 0.6123724356957945f;
    float v;
    switch (t) {
      case 0:  v = 1.0f; break;
      case 1:  v = y; break;
      case 2:  v = z; break;
      case 3:  v = x; break;
      case 4:  v = s3 * x * y; break;
      case 5:  v = s3 * y * z; break;
      case 6:  v = 0.5f * (3.f * z * z - 1.f); break;
      case 7:  v = s3 * x * z; break;
      case 8:  v = 0.5f * s3 * (x * x - y * y); break;
      case 9:  v = c1 * y * (3.f * x * x - y * y); break;
      case 10: v = s15 * x * y * z; break;
      case 11: v = c2 * y * (5.f * z * z - 1.f); break;
      case 12: v = 0.5f * z * (5.f * z * z - 3.f); break;
      case 13: v = c2 * x * (5.f * z * z - 1.f); break;
      case 14: v = 0.5f * s15 * z * (x * x - y * y); break;
      default: v = c1 * x * (x * x - 3.f * y * y); break;
    }
    sphout[(size_t)p * 16 + t] = v * pm;
  }
}

// ---------------------------------------------------------------------------
// Dominant GEMM, fused with radial contraction:
//   edges_scalar[p,f] = sum_r radial[p,r] * (h1[p,:] . W2[:, r*128+f] + b2[r*128+f])
// Block = 64 edges (4 x 16 M-subtiles), 8 waves x 16 output features.
// Each B fragment is reused by 4 WMMAs (4x less L2 traffic on B); A frags are
// re-read from LDS every iteration (NO_HOIST) so nothing spills to scratch.
// ---------------------------------------------------------------------------
__global__ void gemm1_fused(const _Float16* __restrict__ h1,
                            const _Float16* __restrict__ W2T,  // 1024 cols x 128
                            const float* __restrict__ b2,
                            const float* __restrict__ radial,
                            const unsigned char* __restrict__ pmask,
                            _Float16* __restrict__ es) {
  __shared__ _Float16 As[64 * DIM];  // 16 KB
  __shared__ float Rad[64 * 8];      // 2 KB
  int tid = threadIdx.x;
  int pbase = blockIdx.x * 64;
  {
    const uint4* src = (const uint4*)(h1 + (size_t)pbase * DIM);
    uint4* dst = (uint4*)As;
    for (int c = tid; c < 64 * DIM / 8; c += 256) dst[c] = src[c];
    const uint4* rs = (const uint4*)(radial + (size_t)pbase * 8);
    uint4* rd = (uint4*)Rad;
    for (int c = tid; c < 64 * 8 / 4; c += 256) rd[c] = rs[c];
  }
  __syncthreads();
  int wave = tid >> 5, lane = tid & 31, hi = lane >> 4, lm = lane & 15;
  int fb = wave * 16;
  v8f facc[4] = {{}, {}, {}, {}};
  for (int r = 0; r < 8; ++r) {
    NO_HOIST();  // force per-iteration LDS reads; keep live set ~90 VGPRs
    int col = r * DIM + fb + lm;
    if (r < 7) {  // prefetch next r's B column into near cache
      __builtin_prefetch(&W2T[(size_t)(col + DIM) * DIM], 0, 3);
      __builtin_prefetch(&W2T[(size_t)(col + DIM) * DIM + 64], 0, 3);
    }
    v8f acc[4] = {{}, {}, {}, {}};
#pragma unroll
    for (int kb = 0; kb < DIM; kb += 32) {
      v16h b;
      const _Float16* bp = &W2T[(size_t)col * DIM + kb + hi * 8];
#pragma unroll
      for (int u = 0; u < 8; ++u) { b[u] = bp[u]; b[8 + u] = bp[16 + u]; }
#pragma unroll
      for (int mt = 0; mt < 4; ++mt) {
        v16h a;
        const _Float16* ap = &As[(mt * 16 + lm) * DIM + kb + hi * 8];
#pragma unroll
        for (int u = 0; u < 8; ++u) { a[u] = ap[u]; a[8 + u] = ap[16 + u]; }
        acc[mt] = __builtin_amdgcn_wmma_f32_16x16x32_f16(false, a, false, b,
                                                         (short)0, acc[mt],
                                                         false, false);
      }
      NO_HOIST();  // recycle fragment registers between kb steps
    }
    float bias = b2[col];
#pragma unroll
    for (int mt = 0; mt < 4; ++mt)
#pragma unroll
      for (int jv = 0; jv < 8; ++jv) {
        int m = mt * 16 + jv + hi * 8;
        facc[mt][jv] += Rad[m * 8 + r] * (acc[mt][jv] + bias);
      }
  }
#pragma unroll
  for (int mt = 0; mt < 4; ++mt)
#pragma unroll
    for (int jv = 0; jv < 8; ++jv) {
      int m = mt * 16 + jv + hi * 8;
      int p = pbase + m;
      float v = facc[mt][jv] * (pmask[p] ? 1.0f : 0.0f);
      es[(size_t)p * DIM + fb + lm] = (_Float16)v;
    }
}

// ---------------------------------------------------------------------------
// msg = es @ edge_dense_W (waves 0-7) and cl = es @ sph_dense_W (waves 8-9),
// with fused segment-sum scatter (atomics) into node buffers.
// Block = 320 threads (10 waves), 64 edges (4 M-subtiles, B reused x4).
// ---------------------------------------------------------------------------
__global__ void gemm2_scatter(const _Float16* __restrict__ es,
                              const _Float16* __restrict__ WeT,  // 128 cols x 128
                              const _Float16* __restrict__ WsT,  // 32 cols x 128
                              const float* __restrict__ sphv,
                              const int* __restrict__ ii,
                              float* __restrict__ node_up,
                              float* __restrict__ node_sph) {
  __shared__ _Float16 As[64 * DIM];  // 16 KB
  __shared__ float Sph[64 * 16];     // 4 KB
  __shared__ int Idx[64];
  int tid = threadIdx.x;  // 0..319
  int pbase = blockIdx.x * 64;
  {
    const uint4* src = (const uint4*)(es + (size_t)pbase * DIM);
    uint4* dst = (uint4*)As;
    for (int c = tid; c < 64 * DIM / 8; c += 320) dst[c] = src[c];
    const uint4* ss = (const uint4*)(sphv + (size_t)pbase * 16);
    uint4* sd = (uint4*)Sph;
    for (int c = tid; c < 64 * 16 / 4; c += 320) sd[c] = ss[c];
  }
  if (tid < 64) Idx[tid] = ii[pbase + tid];
  __syncthreads();
  int wave = tid >> 5, lane = tid & 31, hi = lane >> 4, lm = lane & 15;
  const _Float16* Bt = (wave < 8) ? WeT : WsT;
  int col = (wave < 8) ? (wave * 16 + lm) : ((wave - 8) * 16 + lm);
  v8f acc[4] = {{}, {}, {}, {}};
#pragma unroll
  for (int kb = 0; kb < DIM; kb += 32) {
    v16h b;
    const _Float16* bp = &Bt[(size_t)col * DIM + kb + hi * 8];
#pragma unroll
    for (int u = 0; u < 8; ++u) { b[u] = bp[u]; b[8 + u] = bp[16 + u]; }
#pragma unroll
    for (int mt = 0; mt < 4; ++mt) {
      v16h a;
      const _Float16* ap = &As[(mt * 16 + lm) * DIM + kb + hi * 8];
#pragma unroll
      for (int u = 0; u < 8; ++u) { a[u] = ap[u]; a[8 + u] = ap[16 + u]; }
      acc[mt] = __builtin_amdgcn_wmma_f32_16x16x32_f16(false, a, false, b,
                                                       (short)0, acc[mt],
                                                       false, false);
    }
    NO_HOIST();  // keep fragment lifetimes short across unrolled kb steps
  }
  if (wave < 8) {
#pragma unroll
    for (int mt = 0; mt < 4; ++mt)
#pragma unroll
      for (int jv = 0; jv < 8; ++jv) {
        int m = mt * 16 + jv + hi * 8;
        atomicAdd(&node_up[(size_t)Idx[m] * DIM + col], acc[mt][jv]);
      }
  } else {
    int l = col >> 3, s = col & 7;
    int base = l * l, cnt = 2 * l + 1;
#pragma unroll
    for (int mt = 0; mt < 4; ++mt)
#pragma unroll
      for (int jv = 0; jv < 8; ++jv) {
        int m = mt * 16 + jv + hi * 8;
        float v = acc[mt][jv];
        size_t nb = (size_t)Idx[m] * DIM;
        for (int k = 0; k < cnt; ++k) {
          int msph = base + k;
          atomicAdd(&node_sph[nb + msph * 8 + s], v * Sph[m * 16 + msph]);
        }
      }
  }
}

// ---------------------------------------------------------------------------
// Generic WMMA GEMM: C[MxNc] = act(A[MxK] @ B + bias) (+ resid).
// A: f32 row-major (converted to f16 in LDS). Bt: f16 col-major (Nc x K).
// Block = 64 rows (4 M-subtiles, B reused x4), 8 waves x 16-col tiles.
// grid.x = M/64, grid.y = Nc/128.
// ---------------------------------------------------------------------------
__global__ void gemm_f16(const float* __restrict__ A, const _Float16* __restrict__ Bt,
                         const float* __restrict__ bias, const float* __restrict__ resid,
                         float* __restrict__ C, int M, int K, int Nc, int act) {
  __shared__ _Float16 As[64 * 256];  // 32 KB
  int tid = threadIdx.x;
  int mbase = blockIdx.x * 64;
  {
    const float4* src = (const float4*)(A + (size_t)mbase * K);
    for (int c = tid; c < 64 * K / 4; c += 256) {
      float4 f = src[c];
      _Float16 h4[4] = {(_Float16)f.x, (_Float16)f.y, (_Float16)f.z, (_Float16)f.w};
      *(uint2*)(As + c * 4) = *(uint2*)h4;
    }
  }
  __syncthreads();
  int wave = tid >> 5, lane = tid & 31, hi = lane >> 4, lm = lane & 15;
  int ncol = blockIdx.y * 128 + wave * 16 + lm;
  v8f acc[4] = {{}, {}, {}, {}};
  for (int kb = 0; kb < K; kb += 32) {
    NO_HOIST();
    v16h b;
    const _Float16* bp = &Bt[(size_t)ncol * K + kb + hi * 8];
#pragma unroll
    for (int u = 0; u < 8; ++u) { b[u] = bp[u]; b[8 + u] = bp[16 + u]; }
#pragma unroll
    for (int mt = 0; mt < 4; ++mt) {
      v16h a;
      const _Float16* ap = &As[(mt * 16 + lm) * K + kb + hi * 8];
#pragma unroll
      for (int u = 0; u < 8; ++u) { a[u] = ap[u]; a[8 + u] = ap[16 + u]; }
      acc[mt] = __builtin_amdgcn_wmma_f32_16x16x32_f16(false, a, false, b,
                                                       (short)0, acc[mt],
                                                       false, false);
    }
  }
  float bv = bias ? bias[ncol] : 0.0f;
#pragma unroll
  for (int mt = 0; mt < 4; ++mt)
#pragma unroll
    for (int jv = 0; jv < 8; ++jv) {
      int row = mbase + mt * 16 + jv + hi * 8;
      float v = acc[mt][jv] + bv;
      if (act) v = silu_f(v);
      if (resid) v += resid[(size_t)row * Nc + ncol];
      C[(size_t)row * Nc + ncol] = v;
    }
}

// ---------------------------------------------------------------------------
// LayerNorm over D=128 with node mask. One wave (32 lanes) per row.
// ---------------------------------------------------------------------------
__global__ void layernorm_mask(const float* __restrict__ X, const float* __restrict__ sc,
                               const float* __restrict__ bi,
                               const unsigned char* __restrict__ nmask,
                               float* __restrict__ Y) {
  int warp = threadIdx.x >> 5, lane = threadIdx.x & 31;
  int row = blockIdx.x * 8 + warp;
  const float* xr = X + (size_t)row * DIM;
  float v[4];
  float s = 0.f, s2 = 0.f;
#pragma unroll
  for (int t = 0; t < 4; ++t) {
    v[t] = xr[lane + 32 * t];
    s += v[t];
    s2 += v[t] * v[t];
  }
#pragma unroll
  for (int o = 16; o > 0; o >>= 1) {
    s += __shfl_xor(s, o, 32);
    s2 += __shfl_xor(s2, o, 32);
  }
  float mu = s / 128.f;
  float var = s2 / 128.f - mu * mu;
  float inv = rsqrtf(var + 1e-6f);
  float mk = nmask[row] ? 1.0f : 0.0f;
  float* yr = Y + (size_t)row * DIM;
#pragma unroll
  for (int t = 0; t < 4; ++t) {
    int c = lane + 32 * t;
    yr[c] = ((v[t] - mu) * inv * sc[c] + bi[c]) * mk;
  }
}

// ---------------------------------------------------------------------------
// Tensor-decomposition contraction + degree-wise norms:
//   tmp[m,t] = sum_s ns[m,s]*td[SEG[m]][s,t]; trace[l,t] = sum_{m in l} tmp^2
//   out[l*8+t] = sqrt(max(trace,1e-24)) * (2l+1)^0.25
// One wave per node; lane = (l,t).
// ---------------------------------------------------------------------------
__global__ void sph_norms(const float* __restrict__ node_sph, const float* __restrict__ td,
                          const unsigned char* __restrict__ nmask,
                          float* __restrict__ outp) {
  int node = blockIdx.x * 8 + (threadIdx.x >> 5);
  int lt = threadIdx.x & 31;
  int l = lt >> 3, t = lt & 7;
  float mk = nmask[node] ? 1.0f : 0.0f;
  const float* ns = node_sph + (size_t)node * DIM;
  float tr = 0.f;
  int base = l * l, cnt = 2 * l + 1;
  for (int k = 0; k < cnt; ++k) {
    int m = base + k;
    float tmp = 0.f;
#pragma unroll
    for (int s = 0; s < 8; ++s) tmp += ns[m * 8 + s] * mk * td[l * 64 + s * 8 + t];
    tr += tmp * tmp;
  }
  const float LF[4] = {1.0f, 1.3160740129524924f, 1.4953487812212205f, 1.6265765616977853f};
  outp[(size_t)node * 32 + l * 8 + t] = sqrtf(fmaxf(tr, 1e-24f)) * LF[l];
}

// ---------------------------------------------------------------------------
// Final ->1 head layers via shuffle reductions; writes (N,3) output.
// ---------------------------------------------------------------------------
__global__ void heads_final(const float* __restrict__ e2, const float* __restrict__ We3,
                            const float* __restrict__ be3, const float* __restrict__ c1,
                            const float* __restrict__ Wc2, const float* __restrict__ bc2,
                            const float* __restrict__ s1, const float* __restrict__ Ws2,
                            const float* __restrict__ bs2,
                            const unsigned char* __restrict__ nmask,
                            float* __restrict__ outp) {
  int node = blockIdx.x * 8 + (threadIdx.x >> 5);
  int lane = threadIdx.x & 31;
  float ae = 0.f, aq = 0.f, as = 0.f;
#pragma unroll
  for (int t = 0; t < 4; ++t) {
    int c = lane + 32 * t;
    ae += e2[(size_t)node * 128 + c] * We3[c];
  }
#pragma unroll
  for (int t = 0; t < 8; ++t) {
    int c = lane + 32 * t;
    aq += c1[(size_t)node * 256 + c] * Wc2[c];
    as += s1[(size_t)node * 256 + c] * Ws2[c];
  }
#pragma unroll
  for (int o = 16; o > 0; o >>= 1) {
    ae += __shfl_xor(ae, o, 32);
    aq += __shfl_xor(aq, o, 32);
    as += __shfl_xor(as, o, 32);
  }
  if (lane == 0) {
    float mk = nmask[node] ? 1.0f : 0.0f;
    outp[(size_t)node * 3 + 0] = (ae + be3[0]) * mk;
    outp[(size_t)node * 3 + 1] = (aq + bc2[0]) * mk;
    outp[(size_t)node * 3 + 2] = (as + bs2[0]) * mk;
  }
}

// ---------------------------------------------------------------------------
// Workspace layout (bytes)
// ---------------------------------------------------------------------------
#define F16S ((size_t)2)
#define F32S ((size_t)4)
constexpr size_t OFF_W2T   = 0;
constexpr size_t OFF_EDGET = OFF_W2T   + (size_t)1024 * 128 * F16S;
constexpr size_t OFF_SPHT  = OFF_EDGET + (size_t)128 * 128 * F16S;
constexpr size_t OFF_U1A   = OFF_SPHT  + (size_t)32 * 128 * F16S;
constexpr size_t OFF_U1B   = OFF_U1A   + (size_t)256 * 128 * F16S;
constexpr size_t OFF_U1C   = OFF_U1B   + (size_t)128 * 256 * F16S;
constexpr size_t OFF_U1D   = OFF_U1C   + (size_t)256 * 128 * F16S;
constexpr size_t OFF_U2A   = OFF_U1D   + (size_t)128 * 256 * F16S;
constexpr size_t OFF_U2B   = OFF_U2A   + (size_t)256 * 32 * F16S;
constexpr size_t OFF_U2C   = OFF_U2B   + (size_t)128 * 256 * F16S;
constexpr size_t OFF_U2D   = OFF_U2C   + (size_t)256 * 128 * F16S;
constexpr size_t OFF_E1T   = OFF_U2D   + (size_t)128 * 256 * F16S;
constexpr size_t OFF_E2T   = OFF_E1T   + (size_t)128 * 128 * F16S;
constexpr size_t OFF_C1T   = OFF_E2T   + (size_t)128 * 128 * F16S;
constexpr size_t OFF_S1T   = OFF_C1T   + (size_t)256 * 128 * F16S;
constexpr size_t OFF_H1    = ((OFF_S1T + (size_t)256 * 128 * F16S + 255) & ~(size_t)255);
constexpr size_t OFF_RAD   = OFF_H1   + (size_t)N_PAIRS * 128 * F16S;
constexpr size_t OFF_SPHV  = OFF_RAD  + (size_t)N_PAIRS * 8 * F32S;
constexpr size_t OFF_ES    = OFF_SPHV + (size_t)N_PAIRS * 16 * F32S;
constexpr size_t OFF_NUP   = OFF_ES   + (size_t)N_PAIRS * 128 * F16S;
constexpr size_t OFF_NSPH  = OFF_NUP  + (size_t)N_NODES * 128 * F32S;
constexpr size_t OFF_T1    = OFF_NSPH + (size_t)N_NODES * 128 * F32S;
constexpr size_t OFF_T2    = OFF_T1   + (size_t)N_NODES * 256 * F32S;
constexpr size_t OFF_X     = OFF_T2   + (size_t)N_NODES * 128 * F32S;
constexpr size_t OFF_U2IN  = OFF_X    + (size_t)N_NODES * 128 * F32S;
constexpr size_t OFF_E1    = OFF_U2IN + (size_t)N_NODES * 32 * F32S;
constexpr size_t OFF_E2    = OFF_E1   + (size_t)N_NODES * 128 * F32S;
constexpr size_t OFF_C1    = OFF_E2   + (size_t)N_NODES * 128 * F32S;
constexpr size_t OFF_S1    = OFF_C1   + (size_t)N_NODES * 256 * F32S;

extern "C" void kernel_launch(void* const* d_in, const int* in_sizes, int n_in,
                              void* d_out, int out_size, void* d_ws, size_t ws_size,
                              hipStream_t stream) {
  (void)in_sizes; (void)n_in; (void)out_size; (void)ws_size;
  const float* R_ij = (const float*)d_in[0];
  const int* ii = (const int*)d_in[1];
  const int* jj = (const int*)d_in[2];
  const int* Zi = (const int*)d_in[3];
  const unsigned char* pmask = (const unsigned char*)d_in[4];
  const unsigned char* nmask = (const unsigned char*)d_in[5];
  const float* embed = (const float*)d_in[6];
  const float* rc1_W1 = (const float*)d_in[7];
  const float* rc1_b1 = (const float*)d_in[8];
  const float* rc1_W2 = (const float*)d_in[9];
  const float* rc1_b2 = (const float*)d_in[10];
  const float* edgeW = (const float*)d_in[11];
  const float* u1_Wa = (const float*)d_in[12];
  const float* u1_ba = (const float*)d_in[13];
  const float* u1_Wb = (const float*)d_in[14];
  const float* u1_bb = (const float*)d_in[15];
  const float* u1_l1s = (const float*)d_in[16];
  const float* u1_l1b = (const float*)d_in[17];
  const float* u1_Wc = (const float*)d_in[18];
  const float* u1_bc = (const float*)d_in[19];
  const float* u1_Wd = (const float*)d_in[20];
  const float* u1_bd = (const float*)d_in[21];
  const float* u1_l2s = (const float*)d_in[22];
  const float* u1_l2b = (const float*)d_in[23];
  const float* sphW = (const float*)d_in[24];
  const float* tdW = (const float*)d_in[25];
  const float* u2_Wa = (const float*)d_in[26];
  const float* u2_ba = (const float*)d_in[27];
  const float* u2_Wb = (const float*)d_in[28];
  const float* u2_bb = (const float*)d_in[29];
  const float* u2_l1s = (const float*)d_in[30];
  const float* u2_l1b = (const float*)d_in[31];
  const float* u2_Wc = (const float*)d_in[32];
  const float* u2_bc = (const float*)d_in[33];
  const float* u2_Wd = (const float*)d_in[34];
  const float* u2_bd = (const float*)d_in[35];
  const float* u2_l2s = (const float*)d_in[36];
  const float* u2_l2b = (const float*)d_in[37];
  const float* en_W1 = (const float*)d_in[38];
  const float* en_b1 = (const float*)d_in[39];
  const float* en_W2 = (const float*)d_in[40];
  const float* en_b2 = (const float*)d_in[41];
  const float* en_W3 = (const float*)d_in[42];
  const float* en_b3 = (const float*)d_in[43];
  const float* ch_W1 = (const float*)d_in[44];
  const float* ch_b1 = (const float*)d_in[45];
  const float* ch_W2 = (const float*)d_in[46];
  const float* ch_b2 = (const float*)d_in[47];
  const float* sc_W1 = (const float*)d_in[48];
  const float* sc_b1 = (const float*)d_in[49];
  const float* sc_W2 = (const float*)d_in[50];
  const float* sc_b2 = (const float*)d_in[51];
  float* out = (float*)d_out;
  char* ws = (char*)d_ws;

  auto LT = [&](const float* W, size_t off, int K, int Nc) {
    convT<<<(K * Nc + 255) / 256, 256, 0, stream>>>(W, (_Float16*)(ws + off), K, Nc);
  };
  // Weight conversion/transposition (f32 -> f16 col-major)
  LT(rc1_W2, OFF_W2T, 128, 1024);
  LT(edgeW, OFF_EDGET, 128, 128);
  LT(sphW, OFF_SPHT, 128, 32);
  LT(u1_Wa, OFF_U1A, 128, 256);
  LT(u1_Wb, OFF_U1B, 256, 128);
  LT(u1_Wc, OFF_U1C, 128, 256);
  LT(u1_Wd, OFF_U1D, 256, 128);
  LT(u2_Wa, OFF_U2A, 32, 256);
  LT(u2_Wb, OFF_U2B, 256, 128);
  LT(u2_Wc, OFF_U2C, 128, 256);
  LT(u2_Wd, OFF_U2D, 256, 128);
  LT(en_W1, OFF_E1T, 128, 128);
  LT(en_W2, OFF_E2T, 128, 128);
  LT(ch_W1, OFF_C1T, 128, 256);
  LT(sc_W1, OFF_S1T, 128, 256);

  // Zero segment-sum accumulators (re-done every call: graph-safe determinism)
  hipMemsetAsync(ws + OFF_NUP, 0, (size_t)N_NODES * 128 * F32S, stream);
  hipMemsetAsync(ws + OFF_NSPH, 0, (size_t)N_NODES * 128 * F32S, stream);

  // Edge geometry + rc1 layer 1
  edge_geom<<<N_PAIRS, 128, 0, stream>>>(R_ij, ii, jj, Zi, pmask, nmask, embed,
                                         rc1_W1, rc1_b1, (_Float16*)(ws + OFF_H1),
                                         (float*)(ws + OFF_RAD), (float*)(ws + OFF_SPHV));
  // Dominant GEMM fused with radial contraction
  gemm1_fused<<<N_PAIRS / 64, 256, 0, stream>>>(
      (const _Float16*)(ws + OFF_H1), (const _Float16*)(ws + OFF_W2T), rc1_b2,
      (const float*)(ws + OFF_RAD), pmask, (_Float16*)(ws + OFF_ES));
  // msg/cl GEMMs + segment-sum scatter
  gemm2_scatter<<<N_PAIRS / 64, 320, 0, stream>>>(
      (const _Float16*)(ws + OFF_ES), (const _Float16*)(ws + OFF_EDGET),
      (const _Float16*)(ws + OFF_SPHT), (const float*)(ws + OFF_SPHV), ii,
      (float*)(ws + OFF_NUP), (float*)(ws + OFF_NSPH));

  dim3 g2(N_NODES / 64, 2), g1(N_NODES / 64, 1);
  float* T1 = (float*)(ws + OFF_T1);
  float* T2 = (float*)(ws + OFF_T2);
  float* X = (float*)(ws + OFF_X);

  // update 1 (input = segment-summed msg; previous x == 0)
  gemm_f16<<<g2, 256, 0, stream>>>((const float*)(ws + OFF_NUP),
                                   (const _Float16*)(ws + OFF_U1A), u1_ba, nullptr,
                                   T1, N_NODES, 128, 256, 1);
  gemm_f16<<<g1, 256, 0, stream>>>(T1, (const _Float16*)(ws + OFF_U1B), u1_bb, nullptr,
                                   T2, N_NODES, 256, 128, 0);
  layernorm_mask<<<N_NODES / 8, 256, 0, stream>>>(T2, u1_l1s, u1_l1b, nmask, X);
  gemm_f16<<<g2, 256, 0, stream>>>(X, (const _Float16*)(ws + OFF_U1C), u1_bc, nullptr,
                                   T1, N_NODES, 128, 256, 1);
  gemm_f16<<<g1, 256, 0, stream>>>(T1, (const _Float16*)(ws + OFF_U1D), u1_bd, X,
                                   T2, N_NODES, 256, 128, 0);
  layernorm_mask<<<N_NODES / 8, 256, 0, stream>>>(T2, u1_l2s, u1_l2b, nmask, X);

  // spherical-tensor contraction -> degree norms
  sph_norms<<<N_NODES / 8, 256, 0, stream>>>((const float*)(ws + OFF_NSPH), tdW, nmask,
                                             (float*)(ws + OFF_U2IN));

  // update 2 (K=32 first layer)
  gemm_f16<<<g2, 256, 0, stream>>>((const float*)(ws + OFF_U2IN),
                                   (const _Float16*)(ws + OFF_U2A), u2_ba, nullptr,
                                   T1, N_NODES, 32, 256, 1);
  gemm_f16<<<g1, 256, 0, stream>>>(T1, (const _Float16*)(ws + OFF_U2B), u2_bb, X,
                                   T2, N_NODES, 256, 128, 0);
  layernorm_mask<<<N_NODES / 8, 256, 0, stream>>>(T2, u2_l1s, u2_l1b, nmask, X);
  gemm_f16<<<g2, 256, 0, stream>>>(X, (const _Float16*)(ws + OFF_U2C), u2_bc, nullptr,
                                   T1, N_NODES, 128, 256, 1);
  gemm_f16<<<g1, 256, 0, stream>>>(T1, (const _Float16*)(ws + OFF_U2D), u2_bd, X,
                                   T2, N_NODES, 256, 128, 0);
  layernorm_mask<<<N_NODES / 8, 256, 0, stream>>>(T2, u2_l2s, u2_l2b, nmask, X);

  // heads
  gemm_f16<<<g1, 256, 0, stream>>>(X, (const _Float16*)(ws + OFF_E1T), en_b1, nullptr,
                                   (float*)(ws + OFF_E1), N_NODES, 128, 128, 1);
  gemm_f16<<<g1, 256, 0, stream>>>((const float*)(ws + OFF_E1),
                                   (const _Float16*)(ws + OFF_E2T), en_b2, nullptr,
                                   (float*)(ws + OFF_E2), N_NODES, 128, 128, 1);
  gemm_f16<<<g2, 256, 0, stream>>>(X, (const _Float16*)(ws + OFF_C1T), ch_b1, nullptr,
                                   (float*)(ws + OFF_C1), N_NODES, 128, 256, 1);
  gemm_f16<<<g2, 256, 0, stream>>>(X, (const _Float16*)(ws + OFF_S1T), sc_b1, nullptr,
                                   (float*)(ws + OFF_S1), N_NODES, 128, 256, 1);
  heads_final<<<N_NODES / 8, 256, 0, stream>>>(
      (const float*)(ws + OFF_E2), en_W3, en_b3, (const float*)(ws + OFF_C1), ch_W2,
      ch_b2, (const float*)(ws + OFF_S1), sc_W2, sc_b2, nmask, out);
}
